// HmmLm_35364760715486
// MI455X (gfx1250) — matrix operand order repaired and checked by reading, not compiled, not run on hardware
//
#include <hip/hip_runtime.h>

#define C_ 1024
#define H_ 256
#define V_ 10000
#define B_ 16
#define N_ 256
#define NWG 16          // scan workgroups; each owns 64 alpha columns

typedef _Float16 h8 __attribute__((ext_vector_type(8)));
typedef __attribute__((ext_vector_type(16))) _Float16 v16h;
typedef __attribute__((ext_vector_type(8)))  float    v8f;

union V16H { v16h v; h8 h[2]; };

// ---------------------------------------------------------------- residual MLP
__global__ __launch_bounds__(256)
void hmm_residual_kernel(const float* __restrict__ emb,
                         const float* __restrict__ W1, const float* __restrict__ b1,
                         const float* __restrict__ W2, const float* __restrict__ b2,
                         const float* __restrict__ g,  const float* __restrict__ beta,
                         float* __restrict__ outLn)
{
    __shared__ float x[H_];
    __shared__ float h1[H_];
    __shared__ float y[H_];
    __shared__ float rbuf[256];
    const int c = blockIdx.x, tid = threadIdx.x;

    x[tid] = emb[c * H_ + tid];
    __syncthreads();
    {
        const float* w = W1 + tid * H_;
        float acc = b1[tid];
        #pragma unroll 4
        for (int k = 0; k < H_; ++k) acc = fmaf(w[k], x[k], acc);
        h1[tid] = fmaxf(acc, 0.f);
    }
    __syncthreads();
    {
        const float* w = W2 + tid * H_;
        float acc = b2[tid];
        #pragma unroll 4
        for (int k = 0; k < H_; ++k) acc = fmaf(w[k], h1[k], acc);
        y[tid] = fmaxf(acc, 0.f) + h1[tid];
    }
    __syncthreads();
    rbuf[tid] = y[tid];
    __syncthreads();
    for (int s = 128; s > 0; s >>= 1) { if (tid < s) rbuf[tid] += rbuf[tid + s]; __syncthreads(); }
    const float mu = rbuf[0] * (1.0f / H_);
    __syncthreads();
    const float d = y[tid] - mu;
    rbuf[tid] = d * d;
    __syncthreads();
    for (int s = 128; s > 0; s >>= 1) { if (tid < s) rbuf[tid] += rbuf[tid + s]; __syncthreads(); }
    const float var = rbuf[0] * (1.0f / H_);
    outLn[c * H_ + tid] = g[tid] * d * rsqrtf(var + 1e-5f) + beta[tid];
}

// ---------------------------------------------------------------- start head + log_softmax
__global__ __launch_bounds__(1024)
void hmm_start_head_kernel(const float* __restrict__ ln, const float* __restrict__ Wo,
                           const float* __restrict__ bo, float* __restrict__ startOut)
{
    __shared__ float rbuf[1024];
    const int tid = threadIdx.x;
    const float* a = ln + tid * H_;
    float acc = bo[0];
    #pragma unroll 4
    for (int k = 0; k < H_; ++k) acc = fmaf(a[k], Wo[k], acc);
    rbuf[tid] = acc;
    __syncthreads();
    for (int s = 512; s > 0; s >>= 1) { if (tid < s) rbuf[tid] = fmaxf(rbuf[tid], rbuf[tid + s]); __syncthreads(); }
    const float m = rbuf[0];
    __syncthreads();
    rbuf[tid] = __expf(acc - m);
    __syncthreads();
    for (int s = 512; s > 0; s >>= 1) { if (tid < s) rbuf[tid] += rbuf[tid + s]; __syncthreads(); }
    const float lse = __logf(rbuf[0]) + m;
    startOut[tid] = acc - lse;
}

// ---------------------------------------------------------------- f32 -> f16
__global__ void hmm_cvt_f16_kernel(const float* __restrict__ src, _Float16* __restrict__ dst, int n)
{
    int i = blockIdx.x * blockDim.x + threadIdx.x;
    if (i < n) dst[i] = (_Float16)src[i];
}

// ---------------------------------------------------------------- generic WMMA GEMM:
// C[M,N] = A[M,K] (f16, K-major) * B[N,K]^T (f16, K-major) + bias[N]
__global__ __launch_bounds__(32)
void hmm_gemm16_kernel(const _Float16* __restrict__ A, const _Float16* __restrict__ Bm,
                       const float* __restrict__ bias, float* __restrict__ Cm,
                       int K, int ldc)
{
    const int n0 = blockIdx.x * 16, m0 = blockIdx.y * 16;
    const int lane = threadIdx.x;
    v8f acc = {};
    const _Float16* Arow = A  + (long)(m0 + (lane & 15)) * K;
    const _Float16* Brow = Bm + (long)(n0 + (lane & 15)) * K;
    const int aoff = (lane < 16) ? 0 : 8;
    const int boff = (lane < 16) ? 0 : 8;
    for (int k0 = 0; k0 < K; k0 += 32) {
        V16H a, b;
        a.h[0] = *(const h8*)(Arow + k0 + aoff);
        a.h[1] = *(const h8*)(Arow + k0 + 16 + aoff);
        b.h[0] = *(const h8*)(Brow + k0 + boff);
        b.h[1] = *(const h8*)(Brow + k0 + 16 + boff);
        acc = __builtin_amdgcn_wmma_f32_16x16x32_f16(false, a.v, false, b.v,
                                                     (short)0, acc, false, false);
    }
    #pragma unroll
    for (int r = 0; r < 8; ++r) {
        const int row = m0 + r + ((lane >> 4) << 3);
        const int col = n0 + (lane & 15);
        Cm[(long)row * ldc + col] = acc[r] + bias[col];
    }
}

// ---------------------------------------------------------------- transition row log_softmax
// -> transposed exp(T) in f16:  expTt[j*C + i] = exp(logT[i][j] - lse_i)
__global__ __launch_bounds__(256)
void hmm_trans_norm_kernel(const float* __restrict__ logT, _Float16* __restrict__ expTt)
{
    __shared__ float rbuf[256];
    const int i = blockIdx.x, tid = threadIdx.x;
    const float* row = logT + (long)i * C_;
    float m = -3.0e38f;
    for (int j = tid; j < C_; j += 256) m = fmaxf(m, row[j]);
    rbuf[tid] = m; __syncthreads();
    for (int s = 128; s > 0; s >>= 1) { if (tid < s) rbuf[tid] = fmaxf(rbuf[tid], rbuf[tid + s]); __syncthreads(); }
    m = rbuf[0]; __syncthreads();
    float sum = 0.f;
    for (int j = tid; j < C_; j += 256) sum += __expf(row[j] - m);
    rbuf[tid] = sum; __syncthreads();
    for (int s = 128; s > 0; s >>= 1) { if (tid < s) rbuf[tid] += rbuf[tid + s]; __syncthreads(); }
    const float lse = __logf(rbuf[0]) + m;
    for (int j = tid; j < C_; j += 256)
        expTt[(long)j * C_ + i] = (_Float16)__expf(row[j] - lse);
}

// ---------------------------------------------------------------- emission row lse over V
__global__ __launch_bounds__(256)
void hmm_emis_lse_kernel(const float* __restrict__ logE, float* __restrict__ lseE)
{
    __shared__ float rbuf[256];
    const int c = blockIdx.x, tid = threadIdx.x;
    const float* row = logE + (long)c * V_;
    float m = -3.0e38f;
    for (int v = tid; v < V_; v += 256) m = fmaxf(m, row[v]);
    rbuf[tid] = m; __syncthreads();
    for (int s = 128; s > 0; s >>= 1) { if (tid < s) rbuf[tid] = fmaxf(rbuf[tid], rbuf[tid + s]); __syncthreads(); }
    m = rbuf[0]; __syncthreads();
    float sum = 0.f;
    for (int v = tid; v < V_; v += 256) sum += __expf(row[v] - m);
    rbuf[tid] = sum; __syncthreads();
    for (int s = 128; s > 0; s >>= 1) { if (tid < s) rbuf[tid] += rbuf[tid + s]; __syncthreads(); }
    if (tid == 0) lseE[c] = __logf(rbuf[0]) + m;
}

// ---------------------------------------------------------------- gather emit log-probs:
// emitLog[(t*B + b)*C + c] = logE[c, text[b,t]] - lseE[c]
__global__ __launch_bounds__(256)
void hmm_emit_gather_kernel(const float* __restrict__ logE, const float* __restrict__ lseE,
                            const int* __restrict__ text, float* __restrict__ emitLog)
{
    const int bt = blockIdx.x;          // bt = t*B + b
    const int t = bt / B_, b = bt % B_;
    const int v = text[b * N_ + t];
    for (int c = threadIdx.x; c < C_; c += 256)
        emitLog[(long)bt * C_ + c] = logE[(long)c * V_ + v] - lseE[c];
}

// ---------------------------------------------------------------- barrier init
__global__ void hmm_init_kernel(unsigned* bar)
{
    if (threadIdx.x < 32) bar[threadIdx.x] = 0u;
}

// ---------------------------------------------------------------- grid barrier
__device__ __forceinline__ void hmm_gridbar(unsigned* cnt, unsigned* gen, unsigned& mygen)
{
    __syncthreads();
    if (threadIdx.x == 0) {
        __threadfence();
        const unsigned g = mygen;
        unsigned arr = __hip_atomic_fetch_add(cnt, 1u, __ATOMIC_ACQ_REL, __HIP_MEMORY_SCOPE_AGENT);
        if (arr == NWG - 1) {
            __hip_atomic_store(cnt, 0u, __ATOMIC_RELAXED, __HIP_MEMORY_SCOPE_AGENT);
            __hip_atomic_fetch_add(gen, 1u, __ATOMIC_RELEASE, __HIP_MEMORY_SCOPE_AGENT);
        } else {
            while (__hip_atomic_load(gen, __ATOMIC_ACQUIRE, __HIP_MEMORY_SCOPE_AGENT) == g)
                __builtin_amdgcn_s_sleep(1);
        }
        mygen = g + 1;
    }
    __syncthreads();
}

// ---------------------------------------------------------------- persistent forward recursion
// 16 WGs x 256 threads. WG owns j in [j0, j0+64): 4 j-tiles of 16.
// 8 waves = (j-tile jt = wv&3) x (K-half kh = wv>>2, 512 each).
__global__ __launch_bounds__(256)
void hmm_scan_kernel(const float* __restrict__ start,
                     const float* __restrict__ emitLog,
                     const _Float16* __restrict__ expTt,
                     float* __restrict__ alpha,
                     unsigned* bar, float* __restrict__ out)
{
    __shared__ __align__(16) _Float16 Ash[16 * 1024];   // exp(alpha - m_b), f16 (32KB)
    __shared__ float red[16][17];
    __shared__ float mrow[16];
    __shared__ float accsh[4][32][8];                    // K-half partials (4KB)

    const int tid = threadIdx.x;
    const int j0  = blockIdx.x * 64;
    unsigned mygen = 0;
    unsigned* cnt  = bar;
    unsigned* genp = bar + 16;

    // alpha0 = start + emit[:,0]   (1024 elements of this WG's slice)
    for (int idx = tid; idx < 16 * 64; idx += 256) {
        const int b = idx >> 6, j = j0 + (idx & 63);
        alpha[b * C_ + j] = start[j] + emitLog[b * C_ + j];
    }
    hmm_gridbar(cnt, genp, mygen);

    const int wv = tid >> 5, lane = tid & 31;
    const int jt = wv & 3;        // which 16-col tile
    const int kh = wv >> 2;       // K half: 0 or 1
    const int mA   = lane & 15;
    const int aoff = (lane < 16) ? 0 : 8;
    const int boff = (lane < 16) ? 0 : 8;
    const _Float16* Brow = expTt + (long)(j0 + jt * 16 + (lane & 15)) * C_;

    for (int t = 1; t < N_; ++t) {
        // ---- build A = exp(alpha - rowmax) in LDS (redundant per WG; 1 barrier/step)
        {
            const int b = tid >> 4, ch = tid & 15;
            const float* arow = alpha + b * C_ + ch * 64;
            float lm = -3.0e38f;
            #pragma unroll 8
            for (int k = 0; k < 64; ++k) lm = fmaxf(lm, arow[k]);
            red[b][ch] = lm;
            __syncthreads();
            if (ch == 0) {
                float m = red[b][0];
                #pragma unroll
                for (int k = 1; k < 16; ++k) m = fmaxf(m, red[b][k]);
                mrow[b] = m;
            }
            __syncthreads();
            const float m = mrow[b];
            _Float16* dst = Ash + b * 1024 + ch * 64;
            #pragma unroll 8
            for (int k = 0; k < 64; ++k) dst[k] = (_Float16)__expf(arow[k] - m);
        }
        __syncthreads();

        // ---- GEMM: this wave does 16 WMMAs over its K half
        v8f acc = {};
        #pragma unroll 4
        for (int kk = 0; kk < 16; ++kk) {
            const int k0 = kh * 512 + kk * 32;
            V16H a, b;
            a.h[0] = *(const h8*)(Ash + mA * 1024 + k0 + aoff);
            a.h[1] = *(const h8*)(Ash + mA * 1024 + k0 + 16 + aoff);
            b.h[0] = *(const h8*)(Brow + k0 + boff);
            b.h[1] = *(const h8*)(Brow + k0 + 16 + boff);
            acc = __builtin_amdgcn_wmma_f32_16x16x32_f16(false, a.v, false, b.v,
                                                         (short)0, acc, false, false);
        }
        if (kh == 1) {
            #pragma unroll
            for (int r = 0; r < 8; ++r) accsh[jt][lane][r] = acc[r];
        }
        __syncthreads();

        // ---- K-half 0 waves reduce + write alpha_next for their tile
        if (kh == 0) {
            #pragma unroll
            for (int r = 0; r < 8; ++r) {
                const float s = acc[r] + accsh[jt][lane][r];
                const int brow = r + ((lane >> 4) << 3);
                const int j = j0 + jt * 16 + (lane & 15);
                alpha[brow * C_ + j] = __logf(fmaxf(s, 1e-30f)) + mrow[brow]
                                     + emitLog[((long)t * B_ + brow) * C_ + j];
            }
        }
        // ---- prefetch next step's emission slice while we wait on the barrier
        if (t + 1 < N_ && tid < 64) {
            const int brow = tid >> 2, chunk = tid & 3;
            __builtin_prefetch(&emitLog[((long)(t + 1) * B_ + brow) * C_ + j0 + chunk * 16], 0, 1);
        }
        hmm_gridbar(cnt, genp, mygen);
    }

    // ---- final: sum_b lse_j alpha[b,j]
    if (blockIdx.x == 0) {
        const int b = tid >> 4, ch = tid & 15;
        const float* arow = alpha + b * C_ + ch * 64;
        float lm = -3.0e38f;
        for (int k = 0; k < 64; ++k) lm = fmaxf(lm, arow[k]);
        red[b][ch] = lm;
        __syncthreads();
        if (ch == 0) {
            float m = red[b][0];
            for (int k = 1; k < 16; ++k) m = fmaxf(m, red[b][k]);
            mrow[b] = m;
        }
        __syncthreads();
        const float m = mrow[b];
        float s = 0.f;
        for (int k = 0; k < 64; ++k) s += __expf(arow[k] - m);
        __syncthreads();
        red[b][ch] = s;
        __syncthreads();
        if (tid == 0) {
            float tot = 0.f;
            for (int bb = 0; bb < 16; ++bb) {
                float ss = 0.f;
                for (int k = 0; k < 16; ++k) ss += red[bb][k];
                tot += __logf(ss) + mrow[bb];
            }
            out[0] = tot;
        }
    }
}

// ================================================================ host
extern "C" void kernel_launch(void* const* d_in, const int* in_sizes, int n_in,
                              void* d_out, int out_size, void* d_ws, size_t ws_size,
                              hipStream_t stream)
{
    (void)in_sizes; (void)n_in; (void)out_size; (void)ws_size;
    const float* start_emb = (const float*)d_in[0];
    const float* state_emb = (const float*)d_in[1];
    const float* pre_emb   = (const float*)d_in[2];
    const float* s_W1 = (const float*)d_in[3],  *s_b1 = (const float*)d_in[4];
    const float* s_W2 = (const float*)d_in[5],  *s_b2 = (const float*)d_in[6];
    const float* s_g  = (const float*)d_in[7],  *s_be = (const float*)d_in[8];
    const float* s_Wo = (const float*)d_in[9],  *s_bo = (const float*)d_in[10];
    const float* t_W1 = (const float*)d_in[11], *t_b1 = (const float*)d_in[12];
    const float* t_W2 = (const float*)d_in[13], *t_b2 = (const float*)d_in[14];
    const float* t_g  = (const float*)d_in[15], *t_be = (const float*)d_in[16];
    const float* t_Wo = (const float*)d_in[17], *t_bo = (const float*)d_in[18];
    const float* e_W1 = (const float*)d_in[19], *e_b1 = (const float*)d_in[20];
    const float* e_W2 = (const float*)d_in[21], *e_b2 = (const float*)d_in[22];
    const float* e_g  = (const float*)d_in[23], *e_be = (const float*)d_in[24];
    const float* e_Wo = (const float*)d_in[25], *e_bo = (const float*)d_in[26];
    const int*   text = (const int*)d_in[27];

    char* base = (char*)d_ws;
    size_t off = 0;
    auto alloc = [&](size_t bytes) -> char* {
        char* p = base + off;
        off += (bytes + 255) & ~(size_t)255;
        return p;
    };
    float*    lnS    = (float*)   alloc((size_t)C_ * H_ * 4);
    float*    lnT    = (float*)   alloc((size_t)C_ * H_ * 4);
    float*    lnE    = (float*)   alloc((size_t)C_ * H_ * 4);
    _Float16* lnT16  = (_Float16*)alloc((size_t)C_ * H_ * 2);
    _Float16* lnE16  = (_Float16*)alloc((size_t)C_ * H_ * 2);
    _Float16* tWo16  = (_Float16*)alloc((size_t)C_ * H_ * 2);
    _Float16* eWo16  = (_Float16*)alloc((size_t)V_ * H_ * 2);
    float*    logT   = (float*)   alloc((size_t)C_ * C_ * 4);
    float*    logE   = (float*)   alloc((size_t)C_ * V_ * 4);
    float*    lseE   = (float*)   alloc((size_t)C_ * 4);
    _Float16* expTt  = (_Float16*)alloc((size_t)C_ * C_ * 2);
    float*    startB = (float*)   alloc((size_t)C_ * 4);
    float*    emitL  = (float*)   alloc((size_t)N_ * B_ * C_ * 4);
    float*    alpha  = (float*)   alloc((size_t)B_ * C_ * 4);
    unsigned* bar    = (unsigned*)alloc(256);

    hmm_init_kernel<<<1, 32, 0, stream>>>(bar);

    hmm_residual_kernel<<<C_, 256, 0, stream>>>(start_emb, s_W1, s_b1, s_W2, s_b2, s_g, s_be, lnS);
    hmm_residual_kernel<<<C_, 256, 0, stream>>>(state_emb, t_W1, t_b1, t_W2, t_b2, t_g, t_be, lnT);
    hmm_residual_kernel<<<C_, 256, 0, stream>>>(pre_emb,   e_W1, e_b1, e_W2, e_b2, e_g, e_be, lnE);

    hmm_start_head_kernel<<<1, 1024, 0, stream>>>(lnS, s_Wo, s_bo, startB);

    hmm_cvt_f16_kernel<<<(C_ * H_ + 255) / 256, 256, 0, stream>>>(lnT,  lnT16, C_ * H_);
    hmm_cvt_f16_kernel<<<(C_ * H_ + 255) / 256, 256, 0, stream>>>(lnE,  lnE16, C_ * H_);
    hmm_cvt_f16_kernel<<<(C_ * H_ + 255) / 256, 256, 0, stream>>>(t_Wo, tWo16, C_ * H_);
    hmm_cvt_f16_kernel<<<(V_ * H_ + 255) / 256, 256, 0, stream>>>(e_Wo, eWo16, V_ * H_);

    // transition logits: (C x H) x (C x H)^T -> C x C
    hmm_gemm16_kernel<<<dim3(C_ / 16, C_ / 16), 32, 0, stream>>>(lnT16, tWo16, t_bo, logT, H_, C_);
    hmm_trans_norm_kernel<<<C_, 256, 0, stream>>>(logT, expTt);

    // emission logits: (C x H) x (V x H)^T -> C x V   (V = 10000 = 625*16)
    hmm_gemm16_kernel<<<dim3(V_ / 16, C_ / 16), 32, 0, stream>>>(lnE16, eWo16, e_bo, logE, H_, V_);
    hmm_emis_lse_kernel<<<C_, 256, 0, stream>>>(logE, lseE);
    hmm_emit_gather_kernel<<<N_ * B_, 256, 0, stream>>>(logE, lseE, text, emitL);

    // persistent cooperative forward scan
    hmm_scan_kernel<<<NWG, 256, 0, stream>>>(startB, emitL, expTt, alpha, bar, (float*)d_out);
}
